// DissipationNN_82849919140383
// MI455X (gfx1250) — compile-verified, matrix operand order checked by reading, not run on hardware
//
#include <hip/hip_runtime.h>
#include <hip/hip_bf16.h>
#include <math.h>

// ---------------------------------------------------------------------------
// DissipationNN on MI455X (gfx1250, wave32, WMMA)
//   - bf16 storage / f32 accumulation via v_wmma_f32_16x16x32_bf16
//   - persistent 16-WGP LSTM scan; per step the shared h operand is DMA'd
//     global->LDS in WMMA-fragment order via global_load_async_to_lds_b128
//     (scope:SCOPE_DEV, tracked by ASYNCcnt), then consumed with ds_load_b128
//   - cluster barrier + device-scope atomic barrier between steps
//   - fused-tanh bf16 WMMA GEMMs for the MLP
//   - VALU L·L^T epilogue (output-bandwidth bound: 67 MB @ 23.3 TB/s)
// ---------------------------------------------------------------------------

typedef __attribute__((ext_vector_type(16))) __bf16        v16bf;
typedef __attribute__((ext_vector_type(8)))  float         v8f;
typedef __attribute__((ext_vector_type(8)))  unsigned int  v8u;

#define B_    8
#define T_    2048
#define D_    32
#define H_    1024
#define W_    1024
#define NUML_ 528
#define NROWS (B_ * T_)        // 16384
#define KT_   (H_ / 32)        // 32 k-tiles in the recurrent GEMM
#define LSTRIDE 12             // words per (kt,lane) fragment slot (8 used + pad)

// ---- helpers --------------------------------------------------------------

__device__ __forceinline__ unsigned short f2bf(float f) {
  unsigned u = __builtin_bit_cast(unsigned, f);
  unsigned r = u + 0x7FFFu + ((u >> 16) & 1u);   // round-to-nearest-even
  return (unsigned short)(r >> 16);
}

// K-pair held by VGPR v / lane-half hi of a 16x32 bf16 A-fragment
// (per ISA 7.12.2: lanes 0-15 VGPR0..3 -> K=0..7, VGPR4..7 -> K=16..23;
//  lanes 16-31 shifted by +8). B assumed symmetric over (N,K).
__device__ __forceinline__ int frag_k(int v, int hi) {
  return ((v >> 2) << 4) + hi * 8 + ((v & 3) << 1);
}

// Load a 16x32 bf16 fragment from a row-major [rows, ldk] bf16 matrix.
// Works for both A (r0 = row tile) and B (r0 = output-column tile, since
// weights are stored [Nout, K] and we compute X @ W^T).
__device__ __forceinline__ v16bf load_frag(const unsigned short* __restrict__ M,
                                           int r0, int ldk, int k0, int lane) {
  int r = r0 + (lane & 15), hi = lane >> 4;
  v8u x;
#pragma unroll
  for (int v = 0; v < 8; ++v) {
    int k = k0 + frag_k(v, hi);
    x[v] = *reinterpret_cast<const unsigned int*>(M + (size_t)r * ldk + k);
  }
  return __builtin_bit_cast(v16bf, x);
}

// A-fragment from the LDS fragment cache (pre-swizzled; 2x ds_load_b128).
__device__ __forceinline__ v16bf load_frag_lds(const unsigned* swz,
                                               int kt, int lane) {
  const unsigned* p = swz + (kt * 32 + lane) * LSTRIDE;
  v8u x;
#pragma unroll
  for (int v = 0; v < 8; ++v) x[v] = p[v];
  return __builtin_bit_cast(v16bf, x);
}

// Build x_t A-fragment (batch rows padded 8->16 with zeros) from fp32 input.
__device__ __forceinline__ v16bf load_frag_x(const float* __restrict__ s,
                                             int t, int lane) {
  int b = lane & 15, hi = lane >> 4;
  v8u x;
#pragma unroll
  for (int v = 0; v < 8; ++v) {
    int k = frag_k(v, hi);                       // always < 32 == D_
    unsigned pk = 0u;
    if (b < B_) {
      const float* p = s + ((size_t)b * T_ + t) * D_ + k;
      pk = (unsigned)f2bf(p[0]) | ((unsigned)f2bf(p[1]) << 16);
    }
    x[v] = pk;
  }
  return __builtin_bit_cast(v16bf, x);
}

__device__ __forceinline__ v8f wmma_bf16(v16bf a, v16bf b, v8f c) {
  return __builtin_amdgcn_wmma_f32_16x16x32_bf16(
      /*neg_a=*/false, a, /*neg_b=*/false, b,
      /*c_mod=*/(short)0, c, /*reuse_a=*/false, /*reuse_b=*/false);
}

__device__ __forceinline__ float sigmoidf_(float x) {
  return 1.f / (1.f + __expf(-x));
}

// ---- setup kernels --------------------------------------------------------

__global__ void cvt_bf16_kernel(const float* __restrict__ src,
                                unsigned short* __restrict__ dst, int n) {
  int i = blockIdx.x * blockDim.x + threadIdx.x;
  if (i < n) dst[i] = f2bf(src[i]);
}

__global__ void init_state_kernel(unsigned short* __restrict__ h_glob,
                                  unsigned int* __restrict__ ctr) {
  int i = blockIdx.x * blockDim.x + threadIdx.x;
  if (i < 16 * H_) h_glob[i] = 0;                // rows 8..15 stay zero forever
  if (i == 0) *ctr = 0u;
}

// ---- LSTM persistent scan (16 workgroups = one cluster) -------------------
// WG i owns h-columns [i*64, i*64+64). Per step:
//   1. async-DMA full padded h [16 x 1024] bf16 global->LDS, pre-swizzled
//      into A-fragment order (device scope -> coherent across WGPs)
//   2. gates[16pad x 256] = x_t@Wih^T + h@Whh^T  (bf16 WMMA, f32 accum)
//   3. elementwise cell update; publish bf16 h-slice
//   4. cluster barrier + device atomic step barrier

__global__ __launch_bounds__(256) void lstm_scan_kernel(
    const float* __restrict__ s_window, const float* __restrict__ b_lstm,
    const unsigned short* __restrict__ wih,   // [4H, D] bf16
    const unsigned short* __restrict__ whh,   // [4H, H] bf16
    unsigned short* __restrict__ h_glob,      // [16, H] bf16 (rows 8..15 = 0)
    unsigned short* __restrict__ hs,          // [B*T, H] bf16, row = b*T + t
    unsigned int* __restrict__ ctr) {
  __shared__ unsigned swz[KT_ * 32 * LSTRIDE];  // h A-fragment cache (48 KB)
  __shared__ float gates_s[16][256];            // [m][gate*64 + j]
  __shared__ float c_s[B_][64];                 // persistent cell state slice

  const int tid = threadIdx.x, lane = tid & 31, wid = tid >> 5;
  const int j0 = blockIdx.x * 64;               // this WG's h-column base
  for (int e = tid; e < B_ * 64; e += 256) c_s[e >> 6][e & 63] = 0.f;
  __syncthreads();

  // wave -> two 16-col gate tiles (16 tiles total: 4 gates x 4 subtiles)
  const int nt0 = wid * 2, nt1 = wid * 2 + 1;
  const int g0 = nt0 >> 2, g1 = nt1 >> 2;
  const int n0 = g0 * H_ + j0 + (nt0 & 3) * 16;   // column into [4H]
  const int n1 = g1 * H_ + j0 + (nt1 & 3) * 16;
  const int c0 = g0 * 64 + (nt0 & 3) * 16 + (lane & 15);
  const int c1 = g1 * 64 + (nt1 & 3) * 16 + (lane & 15);
  const int hi = lane >> 4;

  const unsigned long long hbase = (unsigned long long)h_glob;

  for (int t = 0; t < T_; ++t) {
    // --- 1. stage h: async DMA global -> LDS in fragment order -----------
    // Each 16 B source group (row m, k..k+7, k % 8 == 0) lands on 4
    // contiguous fragment words [kt][lane_t][v0..v0+3]; both ends 16 B
    // aligned, so B128 transfers are legal.
#pragma unroll
    for (int g = tid; g < 16 * (H_ / 8); g += 256) {   // 2048 groups
      int m  = g >> 7;                 // padded batch row 0..15
      int k  = (g & 127) << 3;         // k, multiple of 8
      int kt = k >> 5;
      int lane_t = m + (((k >> 3) & 1) << 4);
      int v0 = ((k >> 4) & 1) << 2;
      unsigned dst = (unsigned)(unsigned long long)(const void*)
                         (&swz[(kt * 32 + lane_t) * LSTRIDE + v0]);
      unsigned off = (unsigned)(m * (H_ * 2) + k * 2);   // byte offset
      asm volatile(
          "global_load_async_to_lds_b128 %0, %1, %2 scope:SCOPE_DEV"
          :: "v"(dst), "v"(off), "s"(hbase) : "memory");
    }
    asm volatile("s_wait_asynccnt 0x0" ::: "memory");
    __syncthreads();

    // --- 2. gate GEMM ------------------------------------------------------
    v8f acc0 = {}, acc1 = {};
    {  // fused input projection (K = 32)
      v16bf ax = load_frag_x(s_window, t, lane);
      acc0 = wmma_bf16(ax, load_frag(wih, n0, D_, 0, lane), acc0);
      acc1 = wmma_bf16(ax, load_frag(wih, n1, D_, 0, lane), acc1);
    }
    for (int kt = 0; kt < KT_; ++kt) {            // recurrent part (K = 1024)
      v16bf ah = load_frag_lds(swz, kt, lane);
      acc0 = wmma_bf16(ah, load_frag(whh, n0, H_, kt * 32, lane), acc0);
      acc1 = wmma_bf16(ah, load_frag(whh, n1, H_, kt * 32, lane), acc1);
    }
#pragma unroll
    for (int v = 0; v < 8; ++v) {
      gates_s[v + 8 * hi][c0] = acc0[v];
      gates_s[v + 8 * hi][c1] = acc1[v];
    }
    __syncthreads();

    // --- 3. cell update (gate order i,f,g,o) -------------------------------
    for (int e = tid; e < B_ * 64; e += 256) {
      int b = e >> 6, j = e & 63, jc = j0 + j;
      float gi = gates_s[b][0 * 64 + j] + b_lstm[0 * H_ + jc];
      float gf = gates_s[b][1 * 64 + j] + b_lstm[1 * H_ + jc];
      float gg = gates_s[b][2 * 64 + j] + b_lstm[2 * H_ + jc];
      float go = gates_s[b][3 * 64 + j] + b_lstm[3 * H_ + jc];
      float cc = sigmoidf_(gf) * c_s[b][j] + sigmoidf_(gi) * tanhf(gg);
      c_s[b][j] = cc;
      unsigned short hb = f2bf(sigmoidf_(go) * tanhf(cc));
      h_glob[(size_t)b * H_ + jc] = hb;
      hs[((size_t)b * T_ + t) * H_ + jc] = hb;
    }
    __threadfence();                              // release h slice to device
    __syncthreads();

    // --- 4. step barrier ---------------------------------------------------
    __builtin_amdgcn_s_cluster_barrier();         // NOP if not cluster-dispatched
    if (tid == 0) {                               // portable 16-WG step barrier
      __hip_atomic_fetch_add(ctr, 1u, __ATOMIC_RELEASE,
                             __HIP_MEMORY_SCOPE_AGENT);
      unsigned tgt = 16u * (unsigned)(t + 1);
      while (__hip_atomic_load(ctr, __ATOMIC_ACQUIRE,
                               __HIP_MEMORY_SCOPE_AGENT) < tgt)
        __builtin_amdgcn_s_sleep(2);
    }
    __syncthreads();
  }
}

// ---- MLP GEMM: Out = act(A @ W^T + bias) ---------------------------------
// 256 threads = 8 waves; wave computes a 16x128 strip (A-frag reused over
// 8 WMMAs). Block = 128x128 tile of the output; B-frags shared across the
// 8 waves stay hot in the read-only WGP$.

template <bool TANH, typename OutT>
__global__ __launch_bounds__(256) void gemm_bias_kernel(
    const unsigned short* __restrict__ A,   // [N, K] bf16
    const unsigned short* __restrict__ Wt,  // [Nout, K] bf16
    const float* __restrict__ bias,         // [Nout]
    OutT* __restrict__ Out,                 // [N, Nout]
    int K, int Nout) {
  const int lane = threadIdx.x & 31, wid = threadIdx.x >> 5;
  const int m0 = blockIdx.x * 128 + wid * 16;
  const int cb = blockIdx.y * 128;

  v8f acc[8] = {};
  for (int k0 = 0; k0 < K; k0 += 32) {
    v16bf a = load_frag(A, m0, K, k0, lane);
#pragma unroll
    for (int ct = 0; ct < 8; ++ct) {
      int n0 = cb + ct * 16;
      if (n0 < Nout)
        acc[ct] = wmma_bf16(a, load_frag(Wt, n0, K, k0, lane), acc[ct]);
    }
  }
  const int hi = lane >> 4;
#pragma unroll
  for (int ct = 0; ct < 8; ++ct) {
    int n0 = cb + ct * 16;
    if (n0 >= Nout) continue;                 // Nout is a multiple of 16
    int col = n0 + (lane & 15);
    float bv = bias[col];
#pragma unroll
    for (int v = 0; v < 8; ++v) {
      int row = m0 + v + 8 * hi;
      float val = acc[ct][v] + bv;
      if (TANH) val = tanhf(val);
      if constexpr (sizeof(OutT) == 2)
        Out[(size_t)row * Nout + col] = f2bf(val);
      else
        Out[(size_t)row * Nout + col] = val;
    }
  }
}

// ---- Cholesky-style PSD assembly: R = L L^T ------------------------------
// One block per row n; output-bandwidth bound, so plain VALU.

__global__ __launch_bounds__(256) void llt_kernel(
    const float* __restrict__ lel, float* __restrict__ out) {
  __shared__ float L[D_][D_ + 1];
  const int n = blockIdx.x, tid = threadIdx.x;
  for (int e = tid; e < D_ * (D_ + 1); e += 256) (&L[0][0])[e] = 0.f;
  __syncthreads();
  for (int idx = tid; idx < NUML_; idx += 256) {
    float x = lel[(size_t)n * NUML_ + idx];
    int r = (int)((sqrtf(8.f * (float)idx + 1.f) - 1.f) * 0.5f);
    while ((r + 1) * (r + 2) / 2 <= idx) ++r;
    while (r * (r + 1) / 2 > idx) --r;
    int c = idx - r * (r + 1) / 2;
    if (r == c) x = fmaxf(x, 0.f) + log1pf(__expf(-fabsf(x)));  // softplus
    L[r][c] = x;
  }
  __syncthreads();
  for (int e = tid; e < D_ * D_; e += 256) {
    int i = e >> 5, j = e & 31, kmax = min(i, j);
    float s = 0.f;
    for (int k = 0; k <= kmax; ++k) s += L[i][k] * L[j][k];
    out[(size_t)n * (D_ * D_) + e] = s;
  }
}

// ---- host-side launch -----------------------------------------------------

extern "C" void kernel_launch(void* const* d_in, const int* in_sizes, int n_in,
                              void* d_out, int out_size, void* d_ws,
                              size_t ws_size, hipStream_t stream) {
  (void)in_sizes; (void)n_in; (void)out_size; (void)ws_size;
  const float* s_window = (const float*)d_in[0];
  const float* w_ih  = (const float*)d_in[1];
  const float* w_hh  = (const float*)d_in[2];
  const float* b_lstm= (const float*)d_in[3];
  const float* w0 = (const float*)d_in[4];  const float* b0 = (const float*)d_in[5];
  const float* w1 = (const float*)d_in[6];  const float* b1 = (const float*)d_in[7];
  const float* w2 = (const float*)d_in[8];  const float* b2 = (const float*)d_in[9];
  const float* w3 = (const float*)d_in[10]; const float* b3 = (const float*)d_in[11];
  float* out = (float*)d_out;

  char* base = (char*)d_ws;
  size_t off = 0;
  auto alloc = [&](size_t bytes) {
    char* p = base + off;
    off = (off + bytes + 255) & ~(size_t)255;
    return p;
  };
  unsigned short* wih_b = (unsigned short*)alloc((size_t)4 * H_ * D_ * 2);
  unsigned short* whh_b = (unsigned short*)alloc((size_t)4 * H_ * H_ * 2);
  unsigned short* w0_b  = (unsigned short*)alloc((size_t)W_ * H_ * 2);
  unsigned short* w1_b  = (unsigned short*)alloc((size_t)W_ * W_ * 2);
  unsigned short* w2_b  = (unsigned short*)alloc((size_t)W_ * W_ * 2);
  unsigned short* w3_b  = (unsigned short*)alloc((size_t)NUML_ * W_ * 2);
  unsigned short* h_glob= (unsigned short*)alloc((size_t)16 * H_ * 2);
  unsigned short* hs    = (unsigned short*)alloc((size_t)NROWS * H_ * 2);
  unsigned short* actA  = (unsigned short*)alloc((size_t)NROWS * W_ * 2);
  unsigned short* actB  = (unsigned short*)alloc((size_t)NROWS * W_ * 2);
  float*          lel   = (float*)alloc((size_t)NROWS * NUML_ * 4);
  unsigned int*   ctr   = (unsigned int*)alloc(256);

  auto cvt = [&](const float* s, unsigned short* d, int n) {
    cvt_bf16_kernel<<<(n + 255) / 256, 256, 0, stream>>>(s, d, n);
  };
  init_state_kernel<<<(16 * H_ + 255) / 256, 256, 0, stream>>>(h_glob, ctr);
  cvt(w_ih, wih_b, 4 * H_ * D_);
  cvt(w_hh, whh_b, 4 * H_ * H_);
  cvt(w0, w0_b, W_ * H_);
  cvt(w1, w1_b, W_ * W_);
  cvt(w2, w2_b, W_ * W_);
  cvt(w3, w3_b, NUML_ * W_);

  // sequential scan: one 16-WGP "cluster", step-barriered
  lstm_scan_kernel<<<16, 256, 0, stream>>>(s_window, b_lstm, wih_b, whh_b,
                                           h_glob, hs, ctr);

  dim3 blk(256);
  dim3 gWW(NROWS / 128, W_ / 128);                    // (128, 8)
  dim3 gWL(NROWS / 128, (NUML_ + 127) / 128);         // (128, 5)
  gemm_bias_kernel<true,  unsigned short><<<gWW, blk, 0, stream>>>(hs,   w0_b, b0, actA, H_, W_);
  gemm_bias_kernel<true,  unsigned short><<<gWW, blk, 0, stream>>>(actA, w1_b, b1, actB, W_, W_);
  gemm_bias_kernel<true,  unsigned short><<<gWW, blk, 0, stream>>>(actB, w2_b, b2, actA, W_, W_);
  gemm_bias_kernel<false, float        ><<<gWL, blk, 0, stream>>>(actA, w3_b, b3, lel,  W_, NUML_);

  llt_kernel<<<NROWS, 256, 0, stream>>>(lel, out);
}
